// GAT_32607391711760
// MI455X (gfx1250) — compile-verified, matrix operand order checked by reading, not run on hardware
//
#include <hip/hip_runtime.h>
#include <math.h>

#define NNODES 50000
#define NEDGES 800000
#define ETOT   (NEDGES + NNODES)   // with self loops
#define FDIM   128                 // F_IN = HEADS*HID = 128
#define NHEADS 4
#define HDIM   32
#define NEG_SLOPE 0.2f

typedef __attribute__((ext_vector_type(16))) __bf16 v16bf;
typedef __attribute__((ext_vector_type(8)))  float  v8f;

// ---------------------------------------------------------------- utilities
__device__ inline void atomicMaxF(float* addr, float val) {
  // float ordering == int ordering for val>=0, == reversed-uint ordering for val<0
  if (val >= 0.0f) atomicMax((int*)addr, __float_as_int(val));
  else             atomicMin((unsigned int*)addr, __float_as_uint(val));
}

// ------------------------------------ fp32 -> bf16 weight prep (TRANSPOSED)
// Stores W^T so the WMMA B-fragment (a column of W) becomes two contiguous
// 16-byte runs -> compiler emits global_load_b128 instead of 16x u16 loads.
__global__ void cvt_w_bf16_t(const float* __restrict__ w1,
                             const float* __restrict__ w2,
                             __bf16* __restrict__ o) {
  int i = blockIdx.x * blockDim.x + threadIdx.x;   // i = k*FDIM + n
  if (i < FDIM * FDIM) {
    int k = i >> 7, n = i & (FDIM - 1);
    o[(size_t)n * FDIM + k]               = (__bf16)w1[i];
    o[FDIM * FDIM + (size_t)n * FDIM + k] = (__bf16)w2[i];
  }
}

// --------------------------------------------------------------- WMMA GEMM
// C[N,128] = A[N,128](f32, cvt->bf16 on load) * B (given as B^T bf16 [N,K]),
// f32 accum. grid.x = N/16 row tiles; 8 waves/block, wave w owns the 16x16
// output tile at col w*16. K=128 consumed in 4 v_wmma_f32_16x16x32_bf16 steps.
__global__ __launch_bounds__(256)
void gemm128_wmma(const float* __restrict__ A,
                  const __bf16* __restrict__ Bt,   // [N=128, K=128] row-major
                  float* __restrict__ C) {
  const int lane = threadIdx.x & 31;
  const int wave = threadIdx.x >> 5;
  const int row0 = blockIdx.x * 16;
  const int col0 = wave * 16;
  const int mr   = lane & 15;            // M (A frag) / N (B frag, C frag)
  const int kb   = (lane >> 4) << 3;     // K base: 0 for lanes 0-15, 8 for 16-31

  v8f acc = {};
  const float*  arow = A  + (size_t)(row0 + mr) * FDIM;
  const __bf16* brow = Bt + (size_t)(col0 + mr) * FDIM;  // column of B = row of B^T

  #pragma unroll
  for (int kt = 0; kt < 4; ++kt) {
    const int k0 = kt * 32;
    v16bf af, bf;
    // A 16x32 bf16 frag: VGPR j<4 -> K = kb + {2j,2j+1}; j>=4 -> +16
    // B 32x16 bf16 frag: lane&15 = N column, same K pattern (contiguous in B^T)
    #pragma unroll
    for (int j = 0; j < 8; ++j) {
      af[j]     = (__bf16)arow[k0 + kb + j];
      af[8 + j] = (__bf16)arow[k0 + 16 + kb + j];
      bf[j]     = brow[k0 + kb + j];
      bf[8 + j] = brow[k0 + 16 + kb + j];
    }
    acc = __builtin_amdgcn_wmma_f32_16x16x32_bf16(
        /*neg_a=*/false, af, /*neg_b=*/false, bf,
        /*c_mod=*/(short)0, acc, /*reuse_a=*/false, /*reuse_b=*/false);
  }
  // C frag: VGPR j -> M = (lane>>4)*8 + j, N = lane&15
  const int rbase = row0 + ((lane >> 4) << 3);
  #pragma unroll
  for (int j = 0; j < 8; ++j)
    C[(size_t)(rbase + j) * FDIM + col0 + mr] = acc[j];
}

// -------------------------------------------- per-node attention logits
__global__ void alphas_kernel(const float* __restrict__ h,
                              const float* __restrict__ a_src,
                              const float* __restrict__ a_dst,
                              float* __restrict__ asrc,
                              float* __restrict__ adst) {
  int i = blockIdx.x * blockDim.x + threadIdx.x;   // i = n*NHEADS + head
  if (i >= NNODES * NHEADS) return;
  int n = i / NHEADS, hh = i % NHEADS;
  const float* hr = h + (size_t)n * FDIM + hh * HDIM;
  float s1 = 0.f, s2 = 0.f;
  #pragma unroll
  for (int d = 0; d < HDIM; ++d) {
    s1 += hr[d] * a_src[hh * HDIM + d];
    s2 += hr[d] * a_dst[hh * HDIM + d];
  }
  asrc[i] = s1; adst[i] = s2;
}

// -------------------------------------------- reset stats + accumulator
__global__ void init_kernel(float* __restrict__ mmax,
                            float* __restrict__ ssum,
                            float* __restrict__ outb) {
  int i = blockIdx.x * blockDim.x + threadIdx.x;
  if (i < NNODES * FDIM) outb[i] = 0.0f;
  if (i < NNODES * NHEADS) {
    mmax[i] = __int_as_float(0xff800000u);   // -inf
    ssum[i] = 0.0f;
  }
}

// -------------------------------------------- pass 1: leaky score + seg-max
__global__ void edge_score_max(const int* __restrict__ esrc,
                               const int* __restrict__ edst,
                               const float* __restrict__ asrc,
                               const float* __restrict__ adst,
                               float* __restrict__ eraw,
                               float* __restrict__ mmax) {
  int e = blockIdx.x * blockDim.x + threadIdx.x;
  if (e >= ETOT) return;
  int s, d;
  if (e < NEDGES) { s = esrc[e]; d = edst[e]; } else { s = d = e - NEDGES; }
  #pragma unroll
  for (int hh = 0; hh < NHEADS; ++hh) {
    float v = asrc[s * NHEADS + hh] + adst[d * NHEADS + hh];
    v = (v > 0.0f) ? v : NEG_SLOPE * v;
    eraw[(size_t)e * NHEADS + hh] = v;
    atomicMaxF(&mmax[d * NHEADS + hh], v);
  }
}

// -------------------------------------------- pass 2: exp + seg-sum
__global__ void edge_expsum(const int* __restrict__ edst,
                            float* __restrict__ eexp,   // in: raw, out: exp
                            const float* __restrict__ mmax,
                            float* __restrict__ ssum) {
  int e = blockIdx.x * blockDim.x + threadIdx.x;
  if (e >= ETOT) return;
  int d = (e < NEDGES) ? edst[e] : (e - NEDGES);
  #pragma unroll
  for (int hh = 0; hh < NHEADS; ++hh) {
    float ex = expf(eexp[(size_t)e * NHEADS + hh] - mmax[d * NHEADS + hh]);
    eexp[(size_t)e * NHEADS + hh] = ex;
    atomicAdd(&ssum[d * NHEADS + hh], ex);
  }
}

// -------------------------------------------- pass 3: weighted scatter-add
// one block per edge, one thread per feature; h/out/ssum stay L2-resident
__global__ __launch_bounds__(128)
void edge_aggregate(const int* __restrict__ esrc,
                    const int* __restrict__ edst,
                    const float* __restrict__ eexp,
                    const float* __restrict__ ssum,
                    const float* __restrict__ h,
                    float* __restrict__ outb) {
  int e = blockIdx.x;
  int f = threadIdx.x;                 // 0..127
  int s, d;
  if (e < NEDGES) { s = esrc[e]; d = edst[e]; } else { s = d = e - NEDGES; }
  int hh = f >> 5;                     // head = f / 32
  const float* hp = h + (size_t)s * FDIM + f;
  __builtin_prefetch(hp, 0, 1);        // -> global_prefetch_b8
  float alpha = eexp[(size_t)e * NHEADS + hh] /
                (ssum[(size_t)d * NHEADS + hh] + 1e-16f);
  atomicAdd(&outb[(size_t)d * FDIM + f], (*hp) * alpha);
}

// -------------------------------------------- bias + relu
__global__ void bias_relu(float* __restrict__ x, const float* __restrict__ b) {
  int i = blockIdx.x * blockDim.x + threadIdx.x;
  if (i >= NNODES * FDIM) return;
  float v = x[i] + b[i & (FDIM - 1)];
  x[i] = (v > 0.0f) ? v : 0.0f;
}

// -------------------------------------------- MLP head: wave per node
__global__ __launch_bounds__(256)
void classifier(const float* __restrict__ h,
                const float* __restrict__ Wc1, const float* __restrict__ bc1,
                const float* __restrict__ Wc2, const float* __restrict__ bc2,
                float* __restrict__ out) {
  int n = blockIdx.x * 8 + (threadIdx.x >> 5);
  if (n >= NNODES) return;
  int j = threadIdx.x & 31;            // wave32: lane = hidden unit
  const float* hr = h + (size_t)n * FDIM;
  float c = bc1[j];
  #pragma unroll 16
  for (int k = 0; k < FDIM; ++k) c += hr[k] * Wc1[k * HDIM + j];
  float t = c * Wc2[j];
  #pragma unroll
  for (int off = 16; off; off >>= 1) t += __shfl_xor(t, off, 32);
  if (j == 0) out[n] = 1.0f / (1.0f + expf(-(t + bc2[0])));
}

// ============================================================== launcher
extern "C" void kernel_launch(void* const* d_in, const int* in_sizes, int n_in,
                              void* d_out, int out_size, void* d_ws, size_t ws_size,
                              hipStream_t stream) {
  const float* x   = (const float*)d_in[0];
  const int*   ei  = (const int*)d_in[1];
  const float* W1  = (const float*)d_in[2];
  const float* as1 = (const float*)d_in[3];
  const float* ad1 = (const float*)d_in[4];
  const float* b1  = (const float*)d_in[5];
  const float* W2  = (const float*)d_in[6];
  const float* as2 = (const float*)d_in[7];
  const float* ad2 = (const float*)d_in[8];
  const float* b2  = (const float*)d_in[9];
  const float* Wc1 = (const float*)d_in[10];
  const float* bc1 = (const float*)d_in[11];
  const float* Wc2 = (const float*)d_in[12];
  const float* bc2 = (const float*)d_in[13];
  float* out = (float*)d_out;

  // workspace layout (floats)
  float* ws   = (float*)d_ws;
  float* hbuf = ws;                                  // [N,128]
  float* outb = hbuf + (size_t)NNODES * FDIM;        // [N,128] (accum, reused)
  float* asrc = outb + (size_t)NNODES * FDIM;        // [N,H]
  float* adst = asrc + (size_t)NNODES * NHEADS;      // [N,H]
  float* mmax = adst + (size_t)NNODES * NHEADS;      // [N,H]
  float* ssum = mmax + (size_t)NNODES * NHEADS;      // [N,H]
  float* eexp = ssum + (size_t)NNODES * NHEADS;      // [Et,H]
  __bf16* wbf   = (__bf16*)(eexp + (size_t)ETOT * NHEADS);
  __bf16* w1bft = wbf;                               // W1^T bf16 [128,128]
  __bf16* w2bft = wbf + FDIM * FDIM;                 // W2^T bf16 [128,128]

  const int* esrc = ei;                               // edge_index[0]
  const int* edst = ei + NEDGES;                      // edge_index[1]

  const int TB = 256;
  cvt_w_bf16_t<<<(FDIM * FDIM + TB - 1) / TB, TB, 0, stream>>>(W1, W2, wbf);

  // ------- GAT layer 1
  gemm128_wmma<<<NNODES / 16, 256, 0, stream>>>(x, w1bft, hbuf);
  alphas_kernel<<<(NNODES * NHEADS + TB - 1) / TB, TB, 0, stream>>>(hbuf, as1, ad1, asrc, adst);
  init_kernel<<<(NNODES * FDIM + TB - 1) / TB, TB, 0, stream>>>(mmax, ssum, outb);
  edge_score_max<<<(ETOT + TB - 1) / TB, TB, 0, stream>>>(esrc, edst, asrc, adst, eexp, mmax);
  edge_expsum<<<(ETOT + TB - 1) / TB, TB, 0, stream>>>(edst, eexp, mmax, ssum);
  edge_aggregate<<<ETOT, 128, 0, stream>>>(esrc, edst, eexp, ssum, hbuf, outb);
  bias_relu<<<(NNODES * FDIM + TB - 1) / TB, TB, 0, stream>>>(outb, b1);

  // ------- GAT layer 2 (input = outb; accum buffer reused after GEMM reads it)
  gemm128_wmma<<<NNODES / 16, 256, 0, stream>>>(outb, w2bft, hbuf);
  alphas_kernel<<<(NNODES * NHEADS + TB - 1) / TB, TB, 0, stream>>>(hbuf, as2, ad2, asrc, adst);
  init_kernel<<<(NNODES * FDIM + TB - 1) / TB, TB, 0, stream>>>(mmax, ssum, outb);
  edge_score_max<<<(ETOT + TB - 1) / TB, TB, 0, stream>>>(esrc, edst, asrc, adst, eexp, mmax);
  edge_expsum<<<(ETOT + TB - 1) / TB, TB, 0, stream>>>(edst, eexp, mmax, ssum);
  edge_aggregate<<<ETOT, 128, 0, stream>>>(esrc, edst, eexp, ssum, hbuf, outb);
  bias_relu<<<(NNODES * FDIM + TB - 1) / TB, TB, 0, stream>>>(outb, b2);

  // ------- classifier head
  classifier<<<NNODES / 8, 256, 0, stream>>>(outb, Wc1, bc1, Wc2, bc2, out);
}
